// SA_Layer_8813272891496
// MI455X (gfx1250) — matrix operand order repaired and checked by reading, not compiled, not run on hardware
//
#include <hip/hip_runtime.h>

// Problem constants (from reference)
#define B_    8
#define P_    8192
#define CIN_  64
#define K_    32
#define M_    (P_/4)            // 2048
#define NROW_ (B_*M_*K_)        // 524288 rows through the MLP
#define CPAD1_ 96               // 67 channels padded to 3 k-tiles of 32
#define ITERS_ 8                // row tiles (of 128) per GEMM block
#define STG_LD 72               // padded LDS stride (bf16 elems) for D staging

typedef __attribute__((ext_vector_type(16))) __bf16 v16bf;
typedef __attribute__((ext_vector_type(8)))  float  v8f;

struct alignas(16) BF8 { __bf16 h[8]; };   // 16-byte vector of 8 bf16

__device__ __forceinline__ int center_pt(int m) {
    // idx_c = linspace(0, P-1, M).astype(int32)
    return (int)((double)m * (double)(P_ - 1) / (double)(M_ - 1));
}

// ---------------------------------------------------------------- utilities
__global__ void zero_kernel(float* __restrict__ p, int n) {
    int i = blockIdx.x * blockDim.x + threadIdx.x;
    if (i < n) p[i] = 0.0f;
}

// feats (B, 64, P) fp32 -> featsT (B, P, 64) bf16
__global__ __launch_bounds__(256) void transpose_feats_kernel(
        const float* __restrict__ feats, __bf16* __restrict__ featsT) {
    int t = blockIdx.x * 256 + threadIdx.x;    // over B*P
    int p = t & (P_ - 1);
    int b = t >> 13;                           // P = 2^13
    const float* src = feats + (size_t)b * CIN_ * P_ + p;
    __bf16* dst = featsT + ((size_t)b * P_ + p) * CIN_;
#pragma unroll 4
    for (int c = 0; c < CIN_; ++c) dst[c] = (__bf16)src[(size_t)c * P_];
}

// centers = xyz[:, idx_c, :]  -> first output region
__global__ __launch_bounds__(256) void centers_kernel(
        const float* __restrict__ xyz, float* __restrict__ outC) {
    int t = blockIdx.x * 256 + threadIdx.x;    // over B*M
    int m = t & (M_ - 1);
    int b = t >> 11;                           // M = 2^11
    int ic = center_pt(m);
    const float* s = xyz + ((size_t)b * P_ + ic) * 3;
    float* d = outC + (size_t)t * 3;
    d[0] = s[0]; d[1] = s[1]; d[2] = s[2];
}

// KNN: one thread per center, sorted top-32 list in registers (static indexing
// only -> stays in VGPRs), point tiles staged through LDS.
__global__ __launch_bounds__(256) void knn_kernel(
        const float* __restrict__ xyz, int* __restrict__ knn) {
    __shared__ float sx[256], sy[256], sz[256], sn[256];
    int b = blockIdx.x >> 3;                   // M/256 = 8 blocks per batch
    int m = ((blockIdx.x & 7) << 8) + threadIdx.x;
    int ic = center_pt(m);
    const float* cp = xyz + ((size_t)b * P_ + ic) * 3;
    float cx = cp[0], cy = cp[1], cz = cp[2];
    float cn = cx*cx + cy*cy + cz*cz;
    float bd[K_]; int bi[K_];
#pragma unroll
    for (int i = 0; i < K_; ++i) { bd[i] = 3.4e38f; bi[i] = 0; }
    for (int t0 = 0; t0 < P_; t0 += 256) {
        __syncthreads();
        {
            const float* pp = xyz + ((size_t)b * P_ + t0 + threadIdx.x) * 3;
            float x = pp[0], y = pp[1], z = pp[2];
            sx[threadIdx.x] = x; sy[threadIdx.x] = y; sz[threadIdx.x] = z;
            sn[threadIdx.x] = x*x + y*y + z*z;
        }
        __syncthreads();
        for (int j = 0; j < 256; ++j) {
            // same formula as reference: |c|^2 + |p|^2 - 2 c.p
            float d = cn + sn[j] - 2.0f * (cx*sx[j] + cy*sy[j] + cz*sz[j]);
            if (d < bd[K_ - 1]) {
                float cd = d; int ci = t0 + j;
#pragma unroll
                for (int s = 0; s < K_; ++s) {
                    if (cd < bd[s]) {
                        float td = bd[s]; bd[s] = cd; cd = td;
                        int   ti = bi[s]; bi[s] = ci; ci = ti;
                    }
                }
            }
        }
    }
    int* o = knn + ((size_t)b * M_ + m) * K_;
#pragma unroll
    for (int i = 0; i < K_; ++i) o[i] = bi[i];
}

// Build X0[N, 96] bf16: [local_xyz(3) | feats(64) | zero pad(29)]
// one wave per row; lane handles channels {l, l+32, l+64}
__global__ __launch_bounds__(256) void gather_pack_kernel(
        const float* __restrict__ xyz, const __bf16* __restrict__ featsT,
        const int* __restrict__ knn, __bf16* __restrict__ X0) {
    int n    = (blockIdx.x * 256 + threadIdx.x) >> 5;   // row index
    int lane = threadIdx.x & 31;
    int m = (n >> 5) & (M_ - 1);
    int b = n >> 16;                                    // M*K = 2^16
    int p  = knn[n];
    int ic = center_pt(m);
    const float*  pxyz = xyz + ((size_t)b * P_ + p)  * 3;
    const float*  cxyz = xyz + ((size_t)b * P_ + ic) * 3;
    const __bf16* f    = featsT + ((size_t)b * P_ + p) * CIN_;
    __bf16* dst = X0 + (size_t)n * CPAD1_;
#pragma unroll
    for (int t = 0; t < 3; ++t) {
        int c = lane + 32 * t;
        __bf16 v;
        if (c < 3)       v = (__bf16)(pxyz[c] - cxyz[c]);
        else if (c < 67) v = f[c - 3];
        else             v = (__bf16)0.0f;
        dst[c] = v;
    }
}

// ---------------------------------------------------------------- WMMA GEMM
// Y[N, O] = X[N, CPAD] * W^T ; optional on-the-fly BN(scale,shift)+ReLU on X.
// Block = 256 threads (8 waves); each wave owns 16-row tiles x 64 columns.
// gridDim.y selects the 64-wide column group (layer3 uses 2).
// B fragments are preloaded into VGPRs so the WMMA chain issues without LDS
// stalls; each block loops over ITERS_ row tiles; D tiles staged through
// padded LDS so global stores are contiguous b128.
template<int KT, bool HASBN>
__global__ __launch_bounds__(256) void gemm_bn_kernel(
        const __bf16* __restrict__ Xin, const float* __restrict__ Win,
        int CinActual, int O, const float* __restrict__ ss,
        __bf16* __restrict__ Yout,
        float* __restrict__ gsum, float* __restrict__ gsq) {
    constexpr int CPAD = KT * 32;
    __shared__ alignas(16) __bf16 ldsW[64 * CPAD];     // 64-row W slice, bf16
    __shared__ alignas(16) __bf16 stg[8 * 16 * STG_LD];// per-wave D staging
    __shared__ float lsum[64], lsq[64];
    __shared__ float ssS[CPAD], ssB[CPAD];

    int tid = threadIdx.x;
    int colBase = blockIdx.y * 64;
    // Stage this block's 64-row W slice (fp32 -> bf16, zero-pad c >= CinActual)
    for (int i = tid; i < 64 * CPAD; i += 256) {
        int o = i / CPAD, c = i - o * CPAD;
        ldsW[i] = (c < CinActual)
                    ? (__bf16)Win[(size_t)(colBase + o) * CinActual + c]
                    : (__bf16)0.0f;
    }
    if (HASBN && tid < CPAD) { ssS[tid] = ss[tid]; ssB[tid] = ss[CPAD + tid]; }
    if (tid < 64) { lsum[tid] = 0.0f; lsq[tid] = 0.0f; }
    __syncthreads();

    int wave = tid >> 5;
    int lane = tid & 31;
    int ln   = lane & 15;
    int hi   = lane >> 4;                     // lane half (ISA A/B layouts)

    // ---- preload B fragments (32x16 bf16): lane = column, K split by half
    v16bf Breg[KT * 4];
#pragma unroll
    for (int kt = 0; kt < KT; ++kt) {
#pragma unroll
        for (int ot = 0; ot < 4; ++ot) {
            const __bf16* wp = ldsW + (size_t)(ot * 16 + ln) * CPAD + kt * 32 + hi * 8;
            BF8 b0 = *reinterpret_cast<const BF8*>(wp);
            BF8 b1 = *reinterpret_cast<const BF8*>(wp + 16);
            v16bf t;
#pragma unroll
            for (int i = 0; i < 8; ++i) { t[i] = b0.h[i]; t[8 + i] = b1.h[i]; }
            Breg[kt * 4 + ot] = t;
        }
    }

    float s1[4] = {0.f, 0.f, 0.f, 0.f};
    float s2[4] = {0.f, 0.f, 0.f, 0.f};
    __bf16* sw = stg + (size_t)wave * (16 * STG_LD);

    for (int it = 0; it < ITERS_; ++it) {
        int rowTile = (blockIdx.x * ITERS_ + it) * 128 + wave * 16;
        int rowA    = rowTile + ln;

        // ---- A fragments: lane ln holds row, two contiguous 8-ch chunks
        v16bf Afr[KT];
#pragma unroll
        for (int kt = 0; kt < KT; ++kt) {
            const __bf16* ap = Xin + (size_t)rowA * CPAD + kt * 32 + hi * 8;
            BF8 a0 = *reinterpret_cast<const BF8*>(ap);
            BF8 a1 = *reinterpret_cast<const BF8*>(ap + 16);
            v16bf A;
            if (HASBN) {
                int c0 = kt * 32 + hi * 8;
#pragma unroll
                for (int i = 0; i < 8; ++i) {
                    float f0 = fmaxf((float)a0.h[i] * ssS[c0 + i]      + ssB[c0 + i],      0.0f);
                    float f1 = fmaxf((float)a1.h[i] * ssS[c0 + 16 + i] + ssB[c0 + 16 + i], 0.0f);
                    A[i]     = (__bf16)f0;
                    A[8 + i] = (__bf16)f1;
                }
            } else {
#pragma unroll
                for (int i = 0; i < 8; ++i) { A[i] = a0.h[i]; A[8 + i] = a1.h[i]; }
            }
            Afr[kt] = A;
        }

        // ---- MACs: ot inner keeps dependent WMMAs 4 apart
        v8f zero = {};
        v8f acc[4];
#pragma unroll
        for (int ot = 0; ot < 4; ++ot) acc[ot] = zero;
#pragma unroll
        for (int kt = 0; kt < KT; ++kt) {
#pragma unroll
            for (int ot = 0; ot < 4; ++ot) {
                acc[ot] = __builtin_amdgcn_wmma_f32_16x16x32_bf16(
                    false, Afr[kt], false, Breg[kt * 4 + ot],
                    (short)0, acc[ot], false, false);
            }
        }

        // ---- epilogue: stats + stage D tile to LDS row-major (padded)
#pragma unroll
        for (int ot = 0; ot < 4; ++ot) {
            int col = ot * 16 + ln;
#pragma unroll
            for (int v = 0; v < 8; ++v) {
                float y = acc[ot][v];
                s1[ot] += y; s2[ot] += y * y;
                sw[(v + hi * 8) * STG_LD + col] = (__bf16)y;   // C/D layout
            }
        }
        // wave-private LDS: same-wave DS ops are in order, no barrier needed
        int r   = lane >> 1;
        int seg = lane & 1;
        const __bf16* src = sw + r * STG_LD + seg * 32;
        __bf16* dst = Yout + (size_t)(rowTile + r) * O + colBase + seg * 32;
#pragma unroll
        for (int i = 0; i < 4; ++i)
            *reinterpret_cast<uint4*>(dst + i * 8) =
                *reinterpret_cast<const uint4*>(src + i * 8);
    }

    // ---- per-channel stats: one LDS + one global atomic set per block
#pragma unroll
    for (int ot = 0; ot < 4; ++ot) {
        atomicAdd(&lsum[ot * 16 + ln], s1[ot]);
        atomicAdd(&lsq [ot * 16 + ln], s2[ot]);
    }
    __syncthreads();
    if (tid < 64) {
        atomicAdd(&gsum[colBase + tid], lsum[tid]);
        atomicAdd(&gsq [colBase + tid], lsq[tid]);
    }
}

// stats -> scale/shift : scale = g*rsqrt(var+eps), shift = beta - mu*scale
__global__ void make_ss_kernel(const float* __restrict__ gsum,
                               const float* __restrict__ gsq,
                               const float* __restrict__ g,
                               const float* __restrict__ beta,
                               float* __restrict__ ss, int O) {
    int o = blockIdx.x * blockDim.x + threadIdx.x;
    if (o < O) {
        const float invN = 1.0f / (float)NROW_;
        float mu  = gsum[o] * invN;
        float var = gsq[o] * invN - mu * mu;
        float sc  = g[o] * rsqrtf(var + 1e-5f);
        ss[o]     = sc;
        ss[O + o] = beta[o] - mu * sc;
    }
}

// BN3 + ReLU + max over K, write out (B, 128, M)
__global__ __launch_bounds__(128) void maxpool_kernel(
        const __bf16* __restrict__ Y3, const float* __restrict__ ss,
        float* __restrict__ outF) {
    int bm = blockIdx.x;          // over B*M
    int o  = threadIdx.x;         // 0..127
    float sc = ss[o], sh = ss[128 + o];
    const __bf16* y = Y3 + (size_t)bm * K_ * 128 + o;
    float best = -3.4e38f;
#pragma unroll 4
    for (int k = 0; k < K_; ++k) {
        float v = fmaxf((float)y[(size_t)k * 128] * sc + sh, 0.0f);
        best = fmaxf(best, v);
    }
    int b = bm >> 11, m = bm & (M_ - 1);
    outF[((size_t)b * 128 + o) * M_ + m] = best;
}

// ---------------------------------------------------------------- launcher
extern "C" void kernel_launch(void* const* d_in, const int* in_sizes, int n_in,
                              void* d_out, int out_size, void* d_ws, size_t ws_size,
                              hipStream_t stream) {
    const float* xyz   = (const float*)d_in[0];
    const float* feats = (const float*)d_in[1];
    const float* W1  = (const float*)d_in[2];   // conv biases cancel through BN
    const float* g1  = (const float*)d_in[4];
    const float* be1 = (const float*)d_in[5];
    const float* W2  = (const float*)d_in[6];
    const float* g2  = (const float*)d_in[8];
    const float* be2 = (const float*)d_in[9];
    const float* W3  = (const float*)d_in[10];
    const float* g3  = (const float*)d_in[12];
    const float* be3 = (const float*)d_in[13];

    // Workspace layout (~240 MiB; Y3 reuses dead X0/Y1 region)
    char* ws = (char*)d_ws;
    __bf16* featsT = (__bf16*)(ws);                        // 8 MiB
    int*    knn    = (int*)  (ws + ((size_t)8  << 20));    // 2 MiB
    float*  stats  = (float*)(ws + ((size_t)10 << 20));    // 768 floats: [layer][sum|sq][128]
    float*  ssbuf  = stats + 768;                          // 512 floats: ss1(128) ss2(128) ss3(256)
    __bf16* X0 = (__bf16*)(ws + ((size_t)16  << 20));      // 96 MiB  [N,96]
    __bf16* Y1 = (__bf16*)(ws + ((size_t)112 << 20));      // 64 MiB  [N,64]
    __bf16* Y2 = (__bf16*)(ws + ((size_t)176 << 20));      // 64 MiB  [N,64]
    __bf16* Y3 = (__bf16*)(ws + ((size_t)16  << 20));      // 128 MiB [N,128] (over X0/Y1)

    float* outC = (float*)d_out;                  // centers (B,M,3)
    float* outF = outC + (size_t)B_ * M_ * 3;     // out (B,128,M)

    const int gemmBlocksX = NROW_ / (128 * ITERS_);        // 512

    zero_kernel<<<8, 256, 0, stream>>>(stats, 768 + 512);
    transpose_feats_kernel<<<(B_ * P_) / 256, 256, 0, stream>>>(feats, featsT);
    centers_kernel<<<(B_ * M_) / 256, 256, 0, stream>>>(xyz, outC);
    knn_kernel<<<B_ * (M_ / 256), 256, 0, stream>>>(xyz, knn);
    gather_pack_kernel<<<(NROW_ * 32) / 256, 256, 0, stream>>>(xyz, featsT, knn, X0);

    // layer 1: 67 -> 64 (input already packed/padded bf16)
    gemm_bn_kernel<3, false><<<dim3(gemmBlocksX, 1), 256, 0, stream>>>(
        X0, W1, 67, 64, nullptr, Y1, stats + 0, stats + 128);
    make_ss_kernel<<<1, 64, 0, stream>>>(stats + 0, stats + 128, g1, be1, ssbuf + 0, 64);

    // layer 2: 64 -> 64, BN1+ReLU fused into A-fragment load
    gemm_bn_kernel<2, true><<<dim3(gemmBlocksX, 1), 256, 0, stream>>>(
        Y1, W2, 64, 64, ssbuf + 0, Y2, stats + 256, stats + 384);
    make_ss_kernel<<<1, 64, 0, stream>>>(stats + 256, stats + 384, g2, be2, ssbuf + 128, 64);

    // layer 3: 64 -> 128, BN2+ReLU fused; gridDim.y = 2 column groups
    gemm_bn_kernel<2, true><<<dim3(gemmBlocksX, 2), 256, 0, stream>>>(
        Y2, W3, 64, 128, ssbuf + 128, Y3, stats + 512, stats + 640);
    make_ss_kernel<<<1, 128, 0, stream>>>(stats + 512, stats + 640, g3, be3, ssbuf + 256, 128);

    // BN3 + ReLU + max over K -> (B,128,M)
    maxpool_kernel<<<B_ * M_, 128, 0, stream>>>(Y3, ssbuf + 256, outF);
}
